// SkeletonEncoder_82961588290158
// MI455X (gfx1250) — matrix operand order repaired and checked by reading, not compiled, not run on hardware
//
#include <hip/hip_runtime.h>

// ---------------- problem constants ----------------
#define BB      32
#define TT      1024
#define IN_DIM  60
#define DM      256
#define LL      4
#define DS      32
#define DC      4
#define DI      512          // EXP * DM
#define DTR     16           // DM / 16
#define BT      (BB * TT)    // 32768 tokens
#define XDW     128          // padded width of x_dbl (80 -> 128)
#define KMAX    512
#define APAD    8            // LDS row pad (elements) = 16B = 4 DWORDs

#define USE_TDM 1            // stage GEMM A-tile via tensor_load_to_lds

typedef __attribute__((ext_vector_type(16))) __bf16 v16bf;
typedef __attribute__((ext_vector_type(8)))  float  v8f;
typedef unsigned int u32x4 __attribute__((ext_vector_type(4)));
typedef unsigned int u32x8 __attribute__((ext_vector_type(8)));

union FragB16 { v16bf v; uint4 q[2]; };

// ---------------- helpers ----------------
__device__ inline float wave_sum32(float v) {
    v += __shfl_xor(v, 16, 32);
    v += __shfl_xor(v, 8, 32);
    v += __shfl_xor(v, 4, 32);
    v += __shfl_xor(v, 2, 32);
    v += __shfl_xor(v, 1, 32);
    return v;
}

// block of 256 threads (8 waves), shared red[8]
__device__ inline float block_sum256(float v, float* red) {
    int wave = threadIdx.x >> 5, lane = threadIdx.x & 31;
    v = wave_sum32(v);
    if (lane == 0) red[wave] = v;
    __syncthreads();
    float t = (threadIdx.x < 8) ? red[threadIdx.x] : 0.f;
    if (wave == 0) {
        t = wave_sum32(t);
        if (lane == 0) red[0] = t;
    }
    __syncthreads();
    float r = red[0];
    __syncthreads();
    return r;
}

__device__ inline float silu_f(float x) { return x / (1.f + __expf(-x)); }

// ---------------- 1. feature encoder + LayerNorm + ReLU ----------------
__global__ void __launch_bounds__(256)
k_feat_ln(const float* __restrict__ x, const float* __restrict__ few,
          const float* __restrict__ feb, const float* __restrict__ lng,
          const float* __restrict__ lnb, float* __restrict__ hout) {
    __shared__ float xs[IN_DIM];
    __shared__ float red[8];
    long long tok = blockIdx.x;
    int d = threadIdx.x;
    if (d < IN_DIM) xs[d] = x[tok * IN_DIM + d];
    __syncthreads();
    float acc = feb[d];
    #pragma unroll 4
    for (int i = 0; i < IN_DIM; ++i) acc += xs[i] * few[d * IN_DIM + i];
    float mu = block_sum256(acc, red) * (1.f / DM);
    float diff = acc - mu;
    float var = block_sum256(diff * diff, red) * (1.f / DM);
    float hv = diff * rsqrtf(var + 1e-5f) * lng[d] + lnb[d];
    hout[tok * DM + d] = fmaxf(hv, 0.f);
}

// ---------------- 2. rmsnorm -> bf16 ----------------
__global__ void __launch_bounds__(256)
k_rmsnorm_bf16(const float* __restrict__ h, const float* __restrict__ g,
               __bf16* __restrict__ out) {
    __shared__ float red[8];
    long long tok = blockIdx.x;
    int d = threadIdx.x;
    float v = h[tok * DM + d];
    float ms = block_sum256(v * v, red) * (1.f / DM);
    float s = rsqrtf(ms + 1e-6f);
    out[tok * DM + d] = (__bf16)(v * s * g[d]);
}

// ---------------- 3. final rmsnorm -> f32 ----------------
__global__ void __launch_bounds__(256)
k_rmsnorm_f32(const float* __restrict__ h, const float* __restrict__ g,
              float* __restrict__ out) {
    __shared__ float red[8];
    long long tok = blockIdx.x;
    int d = threadIdx.x;
    float v = h[tok * DM + d];
    float ms = block_sum256(v * v, red) * (1.f / DM);
    float s = rsqrtf(ms + 1e-6f);
    out[tok * DM + d] = v * s * g[d];
}

// ---------------- 4. elementwise casts ----------------
__global__ void __launch_bounds__(256)
k_cast_bf16(const float* __restrict__ in, __bf16* __restrict__ out, int n) {
    int i = blockIdx.x * 256 + threadIdx.x;
    if (i < n) out[i] = (__bf16)in[i];
}

// cast (valid_rows, K) f32 -> (total_rows, K) bf16, zero pad rows beyond valid
__global__ void __launch_bounds__(256)
k_cast_pad_bf16(const float* __restrict__ in, __bf16* __restrict__ out,
                int valid_rows, int K, int n_total) {
    int i = blockIdx.x * 256 + threadIdx.x;
    if (i < n_total) {
        int row = i / K;
        out[i] = (row < valid_rows) ? (__bf16)in[i] : (__bf16)0.f;
    }
}

__global__ void __launch_bounds__(256)
k_negexp(const float* __restrict__ in, float* __restrict__ out, int n) {
    int i = blockIdx.x * 256 + threadIdx.x;
    if (i < n) out[i] = -__expf(in[i]);
}

// ---------------- 5. WMMA bf16 GEMM:  C(M,N) = A(M,K) * W(N,K)^T ----------------
// Grid = M/16 blocks; block = 8 waves sharing one 16-row A tile staged in LDS by
// the Tensor Data Mover (tensor_load_to_lds, D# per ISA 08_async_tensor §8).
// TDM pad_enable reproduces the (K+APAD)-element LDS row stride used by the
// ds_load fragment reads (pad 4 DWORDs every K/2 DWORDs).
// Waves split N in 64-wide strips (stride 512). REQUIRES N % 64 == 0,
// K in {256, 512}. Fragment layouts per CDNA5 ISA 7.12.2 (wave32).
__global__ void __launch_bounds__(256)
k_gemm_bf16(const __bf16* __restrict__ A, const __bf16* __restrict__ W,
            float* __restrict__ C, int N, int K, int residual) {
    __shared__ __bf16 Alds[16 * (KMAX + APAD)];
    const int tid  = threadIdx.x;
    const int wave = tid >> 5;
    const int lane = tid & 31;
    const int half = lane >> 4;
    const int l16  = lane & 15;
    const int m0   = blockIdx.x * 16;
    const int ldsw = K + APAD;

#if USE_TDM
    if (wave == 0) {
        unsigned int lds_base = (unsigned int)(uintptr_t)(void*)&Alds[0];
        unsigned long long ga = (unsigned long long)(uintptr_t)(A + (long long)m0 * K);
        // pad_interval code: row = K/2 DWORDs = 1 << (code+1)  -> code = log2(K) - 2
        unsigned int picode = (K == 512) ? 7u : 6u;
        u32x4 g0;
        g0[0] = 1u;                                        // count=1 (valid D#)
        g0[1] = lds_base;                                  // lds_addr (bytes)
        g0[2] = (unsigned int)(ga & 0xFFFFFFFFu);          // global_addr[31:0]
        g0[3] = (unsigned int)((ga >> 32) & 0x1FFFFFFu)    // global_addr[56:32]
              | (2u << 30);                                // type = 2 (image)
        u32x8 g1;
        g1[0] = (1u << 16)                                 // data_size = 2 bytes
              | (1u << 20)                                 // pad_enable
              | (picode << 22)                             // pad_interval
              | (3u << 25);                                // pad_amount = 4 DWORDs (16B)
        g1[1] = ((unsigned int)K & 0xFFFFu) << 16;         // tensor_dim0[15:0] = K
        g1[2] = (16u << 16);                               // dim0[31:16]=0 | tensor_dim1=16
        g1[3] = ((unsigned int)K & 0xFFFFu) << 16;         // dim1[31:16]=0 | tile_dim0 = K
        g1[4] = 16u;                                       // tile_dim1 = 16 | tile_dim2 = 0
        g1[5] = (unsigned int)K;                           // tensor_dim0_stride = K
        g1[6] = 0u;                                        // stride0[47:32] | dim1_stride lo
        g1[7] = 0u;                                        // dim1_stride hi
        asm volatile("tensor_load_to_lds %0, %1" :: "s"(g0), "s"(g1) : "memory");
        __builtin_amdgcn_s_wait_tensorcnt(0);
    }
    __syncthreads();
#else
    // cooperative stage of A tile: 16 rows x K bf16, 8 elems (16B) per chunk
    const int cpr = K >> 3;                 // chunks per row
    for (int c = tid; c < 16 * cpr; c += 256) {
        int row = c / cpr, col8 = (c - row * cpr) << 3;
        *(uint4*)&Alds[row * ldsw + col8] =
            *(const uint4*)(A + (long long)(m0 + row) * K + col8);
    }
    __syncthreads();
#endif

    const __bf16* Afrag = &Alds[l16 * ldsw + half * 8];
    const int bkoff = half * 8;

    for (int nb = wave * 64; nb < N; nb += 512) {
        v8f acc[4];
        #pragma unroll
        for (int j = 0; j < 4; ++j)
            #pragma unroll
            for (int e = 0; e < 8; ++e) acc[j][e] = 0.f;

        for (int k0 = 0; k0 < K; k0 += 32) {
            FragB16 a;
            a.q[0] = *(const uint4*)(Afrag + k0);        // ds_load_b128
            a.q[1] = *(const uint4*)(Afrag + k0 + 16);
            #pragma unroll
            for (int j = 0; j < 4; ++j) {
                const __bf16* Wrow =
                    W + (long long)(nb + j * 16 + l16) * K + k0 + bkoff;
                __builtin_prefetch(Wrow + 32, 0, 0);     // speculative, next k-block
                FragB16 b;
                b.q[0] = *(const uint4*)(Wrow);
                b.q[1] = *(const uint4*)(Wrow + 16);
                acc[j] = __builtin_amdgcn_wmma_f32_16x16x32_bf16(
                    false, a.v, false, b.v, (short)0, acc[j], false, false);
            }
        }
        #pragma unroll
        for (int j = 0; j < 4; ++j) {
            int col = nb + j * 16 + l16;
            #pragma unroll
            for (int r = 0; r < 8; ++r) {
                long long idx = (long long)(m0 + r + half * 8) * N + col;
                if (residual) C[idx] += acc[j][r];
                else          C[idx]  = acc[j][r];
            }
        }
    }
}

// ---------------- 6. causal depthwise conv (DC=4) + silu -> bf16 ----------------
__global__ void __launch_bounds__(256)
k_conv_silu(const float* __restrict__ xz, const float* __restrict__ cw,
            const float* __restrict__ cb, __bf16* __restrict__ xcbf) {
    long long i = (long long)blockIdx.x * 256 + threadIdx.x;   // over B*T*DI
    int d = (int)(i % DI);
    long long bt = i / DI;
    int t = (int)(bt % TT);
    long long brow = bt - t;                                   // b*T
    float acc = cb[d];
    #pragma unroll
    for (int k = 0; k < DC; ++k) {
        int ts = t + k - (DC - 1);
        if (ts >= 0) acc += cw[d * DC + k] * xz[(brow + ts) * (2 * DI) + d];
    }
    xcbf[i] = (__bf16)silu_f(acc);
}

// ---------------- 7. dt_proj (K=16) + softplus -> delta ----------------
__global__ void __launch_bounds__(256)
k_dtproj(const float* __restrict__ xdbl, const float* __restrict__ dtw,
         const float* __restrict__ dtb, float* __restrict__ delta) {
    long long i = (long long)blockIdx.x * 256 + threadIdx.x;   // over B*T*DI
    int d = (int)(i % DI);
    long long tok = i / DI;
    const float* xr = xdbl + tok * XDW;
    float acc = dtb[d];
    #pragma unroll
    for (int r = 0; r < DTR; ++r) acc += xr[r] * dtw[d * DTR + r];
    delta[i] = (acc > 20.f) ? acc : log1pf(__expf(acc));
}

// ---------------- 8. momentum selective scan ----------------
// lane = state s (wave32 == DS), wave = channel, block = 8 channels of one batch
__global__ void __launch_bounds__(256)
k_scan(float* __restrict__ dly, const float* __restrict__ xdbl,
       const __bf16* __restrict__ xcbf, const float* __restrict__ Aneg,
       const float* __restrict__ alpha_l, const float* __restrict__ beta_l) {
    int wave = threadIdx.x >> 5;
    int lane = threadIdx.x & 31;
    int dpb = DI / 8;                 // 64
    int b = blockIdx.x / dpb;
    int d = (blockIdx.x % dpb) * 8 + wave;
    float alpha = alpha_l[0];
    float beta  = beta_l[0];
    float Ads = Aneg[d * DS + lane];
    float hs = 0.f, vs = 0.f;
    long long baseBD = ((long long)b * TT) * DI + d;
    long long baseX  = ((long long)b * TT) * XDW;
    for (int t = 0; t < TT; ++t) {
        long long idx = baseBD + (long long)t * DI;
        float delta = dly[idx];
        float xv = (float)xcbf[idx];
        float Bt = xdbl[baseX + (long long)t * XDW + DTR + lane];
        float Ct = xdbl[baseX + (long long)t * XDW + DTR + DS + lane];
        float dA  = __expf(delta * Ads);
        float dBx = delta * Bt * xv;
        vs = beta * vs + dBx;
        hs = dA * hs + alpha * vs + (1.f - alpha) * dBx;
        float c = wave_sum32(hs * Ct);
        if (lane == 0) dly[idx] = c;
    }
}

// ---------------- 9. gate: y = (y + D*xc) * silu(z) -> bf16 ----------------
__global__ void __launch_bounds__(256)
k_gate(const float* __restrict__ dly, const __bf16* __restrict__ xcbf,
       const float* __restrict__ xz, const float* __restrict__ dsk,
       __bf16* __restrict__ ybf) {
    long long i = (long long)blockIdx.x * 256 + threadIdx.x;   // over B*T*DI
    int d = (int)(i % DI);
    long long tok = i / DI;
    float y = dly[i] + dsk[d] * (float)xcbf[i];
    float z = xz[tok * (2 * DI) + DI + d];
    ybf[i] = (__bf16)(y * silu_f(z));
}

// ---------------- host side ----------------
extern "C" void kernel_launch(void* const* d_in, const int* in_sizes, int n_in,
                              void* d_out, int out_size, void* d_ws, size_t ws_size,
                              hipStream_t stream) {
    (void)in_sizes; (void)n_in; (void)out_size; (void)ws_size;
    const float* x        = (const float*)d_in[0];
    const float* fe_w     = (const float*)d_in[1];
    const float* fe_b     = (const float*)d_in[2];
    const float* ln_g     = (const float*)d_in[3];
    const float* ln_b     = (const float*)d_in[4];
    const float* blk_g    = (const float*)d_in[5];
    const float* in_w     = (const float*)d_in[6];   // (L, 2*DI, DM)
    const float* conv_w   = (const float*)d_in[7];   // (L, DI, DC)
    const float* conv_b   = (const float*)d_in[8];
    const float* xp_w     = (const float*)d_in[9];   // (L, 80, DI)
    const float* dtp_w    = (const float*)d_in[10];  // (L, DI, DTR)
    const float* dtp_b    = (const float*)d_in[11];
    const float* A_log    = (const float*)d_in[12];  // (L, DI, DS)
    const float* D_skip   = (const float*)d_in[13];
    const float* out_w    = (const float*)d_in[14];  // (L, DM, DI)
    const float* alpha    = (const float*)d_in[15];
    const float* beta     = (const float*)d_in[16];
    const float* rms_g    = (const float*)d_in[17];

    char* p = (char*)d_ws;
    auto carve = [&](size_t bytes) -> char* {
        char* r = p; p += (bytes + 255) & ~(size_t)255; return r;
    };
    float*  hbuf  = (float*) carve((size_t)BT * DM * 4);
    __bf16* ubf   = (__bf16*)carve((size_t)BT * DM * 2);
    float*  xz    = (float*) carve((size_t)BT * 2 * DI * 4);
    __bf16* xcbf  = (__bf16*)carve((size_t)BT * DI * 2);
    float*  xdbl  = (float*) carve((size_t)BT * XDW * 4);  // padded 80->128
    float*  dly   = (float*) carve((size_t)BT * DI * 4);   // delta, then y
    __bf16* ybf   = (__bf16*)carve((size_t)BT * DI * 2);
    __bf16* win   = (__bf16*)carve((size_t)2 * DI * DM * 2);
    __bf16* wxp   = (__bf16*)carve((size_t)XDW * DI * 2);  // padded rows
    __bf16* wout  = (__bf16*)carve((size_t)DM * DI * 2);
    float*  Abuf  = (float*) carve((size_t)DI * DS * 4);

    const int EW = BT * DI / 256;   // elementwise grid over (B,T,DI)
    const int GEMM_BLOCKS = BT / 16;

    // 1. feature encoder + LN + ReLU
    k_feat_ln<<<BT, 256, 0, stream>>>(x, fe_w, fe_b, ln_g, ln_b, hbuf);

    for (int l = 0; l < LL; ++l) {
        // per-layer weight prep
        k_cast_bf16<<<(2 * DI * DM) / 256, 256, 0, stream>>>(
            in_w + (size_t)l * 2 * DI * DM, win, 2 * DI * DM);
        k_cast_pad_bf16<<<(XDW * DI) / 256, 256, 0, stream>>>(
            xp_w + (size_t)l * 80 * DI, wxp, 80, DI, XDW * DI);
        k_cast_bf16<<<(DM * DI) / 256, 256, 0, stream>>>(
            out_w + (size_t)l * DM * DI, wout, DM * DI);
        k_negexp<<<(DI * DS) / 256, 256, 0, stream>>>(
            A_log + (size_t)l * DI * DS, Abuf, DI * DS);

        // u = rmsnorm(h) -> bf16
        k_rmsnorm_bf16<<<BT, 256, 0, stream>>>(hbuf, blk_g + (size_t)l * DM, ubf);

        // xz = u @ in_w^T      (N=1024, K=256)
        k_gemm_bf16<<<GEMM_BLOCKS, 256, 0, stream>>>(ubf, win, xz, 2 * DI, DM, 0);

        // xc = silu(conv(x_in)) -> bf16
        k_conv_silu<<<EW, 256, 0, stream>>>(xz, conv_w + (size_t)l * DI * DC,
                                            conv_b + (size_t)l * DI, xcbf);

        // x_dbl = xc @ xp_w^T  (N=128 padded, K=512)
        k_gemm_bf16<<<GEMM_BLOCKS, 256, 0, stream>>>(xcbf, wxp, xdbl, XDW, DI, 0);

        // delta = softplus(dt @ dtp_w^T + b)
        k_dtproj<<<EW, 256, 0, stream>>>(xdbl, dtp_w + (size_t)l * DI * DTR,
                                         dtp_b + (size_t)l * DI, dly);

        // momentum selective scan (dly: delta in, y out)
        k_scan<<<BB * (DI / 8), 256, 0, stream>>>(dly, xdbl, xcbf, Abuf,
                                                  alpha + l, beta + l);

        // gate: y = (y + D*xc) * silu(z) -> bf16
        k_gate<<<EW, 256, 0, stream>>>(dly, xcbf, xz, D_skip + (size_t)l * DI, ybf);

        // h += y @ out_w^T     (N=256, K=512, residual)
        k_gemm_bf16<<<GEMM_BLOCKS, 256, 0, stream>>>(ybf, wout, hbuf, DM, DI, 1);
    }

    // final rmsnorm -> d_out (f32)
    k_rmsnorm_f32<<<BT, 256, 0, stream>>>(hbuf, rms_g, (float*)d_out);
}
